// EdgeExpert_23433341567134
// MI455X (gfx1250) — compile-verified, matrix operand order checked by reading, not run on hardware
//
#include <hip/hip_runtime.h>
#include <hip/hip_bf16.h>

// Types for CDNA5 WMMA f16 16x16x32 (A,B = v16h across 32 lanes, C/D = v8f)
typedef __attribute__((ext_vector_type(16))) _Float16 v16h;
typedef __attribute__((ext_vector_type(8)))  _Float16 v8h;
typedef __attribute__((ext_vector_type(8)))  float    v8f;

union V16 { v16h v; v8h p[2]; _Float16 h[16]; };

static __device__ __forceinline__ v8f wmma16(v16h a, v16h b, v8f c) {
  return __builtin_amdgcn_wmma_f32_16x16x32_f16(false, a, false, b, (short)0, c, false, false);
}

static __device__ __forceinline__ float gelu_exact(float v) {
  return 0.5f * v * (1.0f + erff(v * 0.70710678118654752f));
}

// raw v_sqrt_f32 / v_rsq_f32 (skip libm denorm fixup chains)
static __device__ __forceinline__ float fast_sqrt(float v) { return __builtin_amdgcn_sqrtf(v); }
static __device__ __forceinline__ float fast_rsq(float v)  { return __builtin_amdgcn_rsqf(v); }

// ---------------------------------------------------------------------------
// prep: zero pooled accumulator; convert branch + fusion weights fp32 -> fp16
// wF16 layout: [branch(sobel,lap,grad,spatial)][o:32][c:128]; fusF16: [o:128][c:128]
// ---------------------------------------------------------------------------
__global__ void ee_prep(const float* __restrict__ sob_w, const float* __restrict__ lap_w,
                        const float* __restrict__ grad_w, const float* __restrict__ spa_w,
                        const float* __restrict__ fus_w,
                        _Float16* __restrict__ wF16, _Float16* __restrict__ fusF16,
                        float* __restrict__ pooled) {
  int idx = blockIdx.x * 256 + threadIdx.x;   // 32768 threads
  if (idx < 16384) {
    int br = idx >> 12;
    int r  = idx & 4095;
    const float* src = (br == 0) ? sob_w : (br == 1) ? lap_w : (br == 2) ? grad_w : spa_w;
    wF16[idx] = (_Float16)src[r];
  } else {
    int j = idx - 16384;
    fusF16[j] = (_Float16)fus_w[j];
  }
  if (idx < 1024) pooled[idx] = 0.0f;
}

// ---------------------------------------------------------------------------
// pass1: stencils -> 4 branch GEMMs (WMMA f16) -> LN(32) -> GELU
//        -> f16 features [pixel][128] in ws, + atomic spatial-pool sums
// One wave32 block per (b,h,wbase) 16-pixel tile. 8192 blocks.
// Halo handling: clamped (always-valid) offsets + float mask multiplies, so
// every global load is unconditional (no exec-mask branches).
// ---------------------------------------------------------------------------
__global__ __launch_bounds__(32) void ee_pass1(
    const float* __restrict__ x, const _Float16* __restrict__ wF16,
    const float* __restrict__ sb_b, const float* __restrict__ sb_g, const float* __restrict__ sb_be,
    const float* __restrict__ lp_b, const float* __restrict__ lp_g, const float* __restrict__ lp_be,
    const float* __restrict__ gr_b, const float* __restrict__ gr_g, const float* __restrict__ gr_be,
    const float* __restrict__ sp_b, const float* __restrict__ sp_g, const float* __restrict__ sp_be,
    _Float16* __restrict__ featws, float* __restrict__ pooled) {

  const int l  = threadIdx.x;
  const int hi = l >> 4;          // half-wave select (K packing)
  const int n  = l & 15;          // pixel row (A) / out-channel (B,C)
  const int bid = blockIdx.x;
  const int b   = bid >> 10;
  const int rem = bid & 1023;
  const int h   = rem >> 3;
  const int wbase = (rem & 7) << 4;
  const int w = wbase + n;
  const bool hm = (h > 0), hp = (h < 127);
  const bool wm = (w > 0), wp = (w < 127);

  // clamped neighbor offsets (elements) -- always in-bounds
  const int up = hm ? -128 : 0;
  const int dn = hp ?  128 : 0;
  const int lf = wm ? -1 : 0;
  const int rt = wp ?  1 : 0;
  // zero-masks for SAME (zero) padding
  const float mU = hm ? 1.f : 0.f, mD = hp ? 1.f : 0.f;
  const float mL = wm ? 1.f : 0.f, mR = wp ? 1.f : 0.f;
  const float mUL = mU * mL, mUR = mU * mR, mDL = mD * mL, mDR = mD * mR;

  const int iUL = up + lf, iUC = up, iUR = up + rt;
  const int iCL = lf,               iCR = rt;
  const int iDL = dn + lf, iDC = dn, iDR = dn + rt;

  const float* xb = x + ((size_t)b << 21);       // batch base (scalar)
  const int obase0 = (h << 7) + w;               // per-lane spatial offset

  __shared__ _Float16 ldsF[16 * 128];   // [pixel m][channel c]

  const float* biasA[4] = {sb_b, lp_b, gr_b, sp_b};
  const float* gA[4]    = {sb_g, lp_g, gr_g, sp_g};
  const float* beA[4]   = {sb_be, lp_be, gr_be, sp_be};

  v8f acc[4][2] = {};   // [branch][ntile]

  #pragma unroll
  for (int kt = 0; kt < 4; ++kt) {
    V16 feat[4];        // 0=sobel, 1=lap, 2=grad, 3=x (A-fragments for this K-tile)
    #pragma unroll
    for (int j = 0; j < 16; ++j) {
      const int c = kt * 32 + ((j < 8) ? j : j + 8) + 8 * hi;   // ISA 16-bit A K-mapping
      const int o = (c << 14) + obase0;
      float a00 = xb[o + iUL] * mUL;
      float a01 = xb[o + iUC] * mU;
      float a02 = xb[o + iUR] * mUR;
      float a10 = xb[o + iCL] * mL;
      float a11 = xb[o];
      float a12 = xb[o + iCR] * mR;
      float a20 = xb[o + iDL] * mDL;
      float a21 = xb[o + iDC] * mD;
      float a22 = xb[o + iDR] * mDR;
      float sx  = (a02 - a00) + 2.f * (a12 - a10) + (a22 - a20);
      float sy  = (a20 + 2.f * a21 + a22) - (a00 + 2.f * a01 + a02);
      float lpv = a01 + a10 + a12 + a21 - 4.f * a11;
      float sob = fast_sqrt(sx * sx + sy * sy + 1e-8f);
      float lpf = fabsf(lpv);
      float grd = fast_sqrt(sob * sob + lpf * lpf + 1e-8f);
      feat[0].h[j] = (_Float16)sob;
      feat[1].h[j] = (_Float16)lpf;
      feat[2].h[j] = (_Float16)grd;
      feat[3].h[j] = (_Float16)a11;
    }
    const int ko = kt * 32 + 8 * hi;
    #pragma unroll
    for (int br = 0; br < 4; ++br) {
      const _Float16* wb = wF16 + br * 4096;
      V16 b0, b1;
      b0.p[0] = *(const v8h*)(wb + n * 128 + ko);
      b0.p[1] = *(const v8h*)(wb + n * 128 + ko + 16);
      b1.p[0] = *(const v8h*)(wb + (n + 16) * 128 + ko);
      b1.p[1] = *(const v8h*)(wb + (n + 16) * 128 + ko + 16);
      acc[br][0] = wmma16(feat[br].v, b0.v, acc[br][0]);
      acc[br][1] = wmma16(feat[br].v, b1.v, acc[br][1]);
    }
  }

  // Per-branch epilogue: bias, LayerNorm over 32 channels, GELU, stage + pool
  #pragma unroll
  for (int br = 0; br < 4; ++br) {
    const float bias0 = biasA[br][n],  bias1 = biasA[br][n + 16];
    const float g0 = gA[br][n],        g1 = gA[br][n + 16];
    const float be0 = beA[br][n],      be1 = beA[br][n + 16];
    float sum0 = 0.f, sum1 = 0.f;
    #pragma unroll
    for (int r = 0; r < 8; ++r) {
      float v0 = acc[br][0][r] + bias0;
      float v1 = acc[br][1][r] + bias1;
      float s = v0 + v1;
      float q = v0 * v0 + v1 * v1;
      #pragma unroll
      for (int off = 1; off < 16; off <<= 1) {   // stays within 16-lane half
        s += __shfl_xor(s, off, 32);
        q += __shfl_xor(q, off, 32);
      }
      float mean = s * (1.f / 32.f);
      float var  = q * (1.f / 32.f) - mean * mean;
      float rs   = fast_rsq(var + 1e-5f);
      v0 = gelu_exact((v0 - mean) * rs * g0 + be0);
      v1 = gelu_exact((v1 - mean) * rs * g1 + be1);
      sum0 += v0; sum1 += v1;
      const int m = r + 8 * hi;   // pixel row of C-matrix
      ldsF[m * 128 + br * 32 + n]      = (_Float16)v0;
      ldsF[m * 128 + br * 32 + 16 + n] = (_Float16)v1;
    }
    // pooled partial sums: combine both pixel halves, one atomic per channel
    sum0 += __shfl_xor(sum0, 16, 32);
    sum1 += __shfl_xor(sum1, 16, 32);
    if (l < 16) {
      atomicAdd(pooled + b * 128 + br * 32 + n,      sum0);
      atomicAdd(pooled + b * 128 + br * 32 + 16 + n, sum1);
    }
  }
  __syncthreads();

  // store tile pixel-major to ws (coalesced 16B stores)
  const size_t gbase = (size_t)b * 16384 + (size_t)h * 128 + wbase;
  #pragma unroll
  for (int it = 0; it < 8; ++it) {
    int u  = it * 32 + l;        // 0..255 units of 8 halfs
    int p  = u >> 4;
    int c0 = (u & 15) * 8;
    *(v8h*)(featws + (gbase + p) * 128 + c0) = *(const v8h*)(ldsF + p * 128 + c0);
  }
}

// ---------------------------------------------------------------------------
// pass2: pooled mean -> att1 (128->16) GELU -> att2 (16->128) sigmoid
// ---------------------------------------------------------------------------
__global__ void ee_pass2(const float* __restrict__ pooled,
                         const float* __restrict__ a1w, const float* __restrict__ a1b,
                         const float* __restrict__ a2w, const float* __restrict__ a2b,
                         float* __restrict__ ew) {
  __shared__ float sp[128];
  __shared__ float sa[16];
  const int t = threadIdx.x;   // 128 threads
  for (int b = 0; b < 8; ++b) {
    sp[t] = pooled[b * 128 + t] * (1.0f / 16384.0f);
    __syncthreads();
    if (t < 16) {
      float d = a1b[t];
      for (int c = 0; c < 128; ++c) d += a1w[t * 128 + c] * sp[c];
      sa[t] = gelu_exact(d);
    }
    __syncthreads();
    float d = a2b[t];
    #pragma unroll
    for (int o = 0; o < 16; ++o) d += a2w[t * 16 + o] * sa[o];
    ew[b * 128 + t] = 1.0f / (1.0f + expf(-d));
    __syncthreads();
  }
}

// ---------------------------------------------------------------------------
// pass3: weighted fusion GEMM (128x128, WMMA f16) -> LN(128) -> GELU -> +x
// ---------------------------------------------------------------------------
__global__ __launch_bounds__(32) void ee_pass3(
    const float* __restrict__ x, const _Float16* __restrict__ featws,
    const _Float16* __restrict__ fusF16, const float* __restrict__ ew,
    const float* __restrict__ fus_b, const float* __restrict__ fus_g,
    const float* __restrict__ fus_be, float* __restrict__ out) {

  const int l  = threadIdx.x;
  const int hi = l >> 4;
  const int n  = l & 15;
  const int bid = blockIdx.x;
  const int b   = bid >> 10;
  const int rem = bid & 1023;
  const int h   = rem >> 3;
  const int wbase = (rem & 7) << 4;

  __shared__ float ldsE[128];
  __shared__ float ldsO[128 * 17];   // [channel][pixel], padded vs bank conflicts

  #pragma unroll
  for (int i = 0; i < 4; ++i) ldsE[i * 32 + l] = ew[b * 128 + i * 32 + l];
  __syncthreads();

  // A fragments: pixel row = n, channels per ISA K-mapping, scaled by edge_weight
  const size_t gp = ((size_t)b * 16384 + (size_t)h * 128 + wbase + n) * 128;
  V16 a[4];
  #pragma unroll
  for (int kt = 0; kt < 4; ++kt) {
    const int ko = kt * 32 + 8 * hi;
    a[kt].p[0] = *(const v8h*)(featws + gp + ko);
    a[kt].p[1] = *(const v8h*)(featws + gp + ko + 16);
    #pragma unroll
    for (int j = 0; j < 16; ++j) {
      const int c = kt * 32 + ((j < 8) ? j : j + 8) + 8 * hi;
      a[kt].h[j] = (_Float16)((float)a[kt].h[j] * ldsE[c]);
    }
  }

  v8f acc[8] = {};
  #pragma unroll
  for (int nt = 0; nt < 8; ++nt) {
    #pragma unroll
    for (int kt = 0; kt < 4; ++kt) {
      const int ko = kt * 32 + 8 * hi;
      V16 bf;
      bf.p[0] = *(const v8h*)(fusF16 + (nt * 16 + n) * 128 + ko);
      bf.p[1] = *(const v8h*)(fusF16 + (nt * 16 + n) * 128 + ko + 16);
      acc[nt] = wmma16(a[kt].v, bf.v, acc[nt]);
    }
  }

  float fb[8], fg[8], fbe[8];
  #pragma unroll
  for (int nt = 0; nt < 8; ++nt) {
    fb[nt]  = fus_b[nt * 16 + n];
    fg[nt]  = fus_g[nt * 16 + n];
    fbe[nt] = fus_be[nt * 16 + n];
  }

  #pragma unroll
  for (int r = 0; r < 8; ++r) {
    float s = 0.f, q = 0.f;
    #pragma unroll
    for (int nt = 0; nt < 8; ++nt) {
      float v = acc[nt][r] + fb[nt];
      s += v; q += v * v;
    }
    #pragma unroll
    for (int off = 1; off < 16; off <<= 1) {
      s += __shfl_xor(s, off, 32);
      q += __shfl_xor(q, off, 32);
    }
    float mean = s * (1.f / 128.f);
    float var  = q * (1.f / 128.f) - mean * mean;
    float rs   = fast_rsq(var + 1e-5f);
    const int m = r + 8 * hi;
    #pragma unroll
    for (int nt = 0; nt < 8; ++nt) {
      float v = acc[nt][r] + fb[nt];
      v = (v - mean) * rs * fg[nt] + fbe[nt];
      ldsO[(nt * 16 + n) * 17 + m] = gelu_exact(v);
    }
  }
  __syncthreads();

  // transpose out of LDS, fuse residual, coalesced float4 stores (NCHW)
  const size_t obase = ((size_t)b * 128) * 16384 + (size_t)h * 128 + wbase;
  #pragma unroll
  for (int it = 0; it < 16; ++it) {
    int u = it * 32 + l;          // 0..511 units of 4 floats
    int c = u >> 2;
    int woff = (u & 3) * 4;
    const size_t gidx = obase + (size_t)c * 16384 + woff;
    float4 xv = *(const float4*)(x + gidx);
    float4 r4;
    r4.x = ldsO[c * 17 + woff + 0] + xv.x;
    r4.y = ldsO[c * 17 + woff + 1] + xv.y;
    r4.z = ldsO[c * 17 + woff + 2] + xv.z;
    r4.w = ldsO[c * 17 + woff + 3] + xv.w;
    *(float4*)(out + gidx) = r4;
  }
}

// ---------------------------------------------------------------------------
// d_in (pytree sorted-key order after x):
// 0:x 1:att1_b 2:att1_w 3:att2_b 4:att2_w 5:fus_b 6:fus_beta 7:fus_g 8:fus_w
// 9:grad_b 10:grad_beta 11:grad_g 12:grad_w 13:lap_b 14:lap_beta 15:lap_g
// 16:lap_w 17:sobel_b 18:sobel_beta 19:sobel_g 20:sobel_w 21:spatial_b
// 22:spatial_beta 23:spatial_g 24:spatial_w
// ws usage: [0,4K) pooled f32 | [4K,8K) edge_weight f32 | [8K,40K) branch w f16
//           [40K,72K) fus w f16 | [72K, 72K+32MB) features f16   (~33.6 MB total)
// ---------------------------------------------------------------------------
extern "C" void kernel_launch(void* const* d_in, const int* in_sizes, int n_in,
                              void* d_out, int out_size, void* d_ws, size_t ws_size,
                              hipStream_t stream) {
  (void)in_sizes; (void)n_in; (void)out_size; (void)ws_size;
  const float* x        = (const float*)d_in[0];
  const float* att1_b   = (const float*)d_in[1];
  const float* att1_w   = (const float*)d_in[2];
  const float* att2_b   = (const float*)d_in[3];
  const float* att2_w   = (const float*)d_in[4];
  const float* fus_b    = (const float*)d_in[5];
  const float* fus_beta = (const float*)d_in[6];
  const float* fus_g    = (const float*)d_in[7];
  const float* fus_w    = (const float*)d_in[8];
  const float* grad_b   = (const float*)d_in[9];
  const float* grad_be  = (const float*)d_in[10];
  const float* grad_g   = (const float*)d_in[11];
  const float* grad_w   = (const float*)d_in[12];
  const float* lap_b    = (const float*)d_in[13];
  const float* lap_be   = (const float*)d_in[14];
  const float* lap_g    = (const float*)d_in[15];
  const float* lap_w    = (const float*)d_in[16];
  const float* sob_b    = (const float*)d_in[17];
  const float* sob_be   = (const float*)d_in[18];
  const float* sob_g    = (const float*)d_in[19];
  const float* sob_w    = (const float*)d_in[20];
  const float* spa_b    = (const float*)d_in[21];
  const float* spa_be   = (const float*)d_in[22];
  const float* spa_g    = (const float*)d_in[23];
  const float* spa_w    = (const float*)d_in[24];

  char* ws = (char*)d_ws;
  float*     pooled = (float*)(ws + 0);
  float*     ewp    = (float*)(ws + 4096);
  _Float16*  wF16   = (_Float16*)(ws + 8192);
  _Float16*  fusF16 = (_Float16*)(ws + 8192 + 32768);
  _Float16*  featws = (_Float16*)(ws + 8192 + 65536);
  float* out = (float*)d_out;

  ee_prep<<<128, 256, 0, stream>>>(sob_w, lap_w, grad_w, spa_w, fus_w, wF16, fusF16, pooled);
  ee_pass1<<<8192, 32, 0, stream>>>(x, wF16,
                                    sob_b, sob_g, sob_be,
                                    lap_b, lap_g, lap_be,
                                    grad_b, grad_g, grad_be,
                                    spa_b, spa_g, spa_be,
                                    featws, pooled);
  ee_pass2<<<1, 128, 0, stream>>>(pooled, att1_w, att1_b, att2_w, att2_b, ewp);
  ee_pass3<<<8192, 32, 0, stream>>>(x, featws, fusF16, ewp, fus_b, fus_g, fus_beta, out);
}
